// GridSampleNet_10522669875502
// MI455X (gfx1250) — matrix operand order repaired and checked by reading, not compiled, not run on hardware
//
#include <hip/hip_runtime.h>
#include <hip/hip_fp16.h>
#include <stdint.h>
#include <utility>

typedef __attribute__((ext_vector_type(16))) _Float16 v16h;
typedef __attribute__((ext_vector_type(8)))  float    v8f;
typedef __attribute__((ext_vector_type(4)))  unsigned int u32x4;
typedef __attribute__((ext_vector_type(8)))  int      i32x8;
typedef __attribute__((ext_vector_type(4)))  int      i32x4;

#define DIVUP(a,b) (((a)+(b)-1)/(b))

#if defined(__HIP_DEVICE_COMPILE__) && __has_builtin(__builtin_amdgcn_tensor_load_to_lds) && __has_builtin(__builtin_amdgcn_s_wait_tensorcnt)
#define HAVE_TDM 1
#else
#define HAVE_TDM 0
#endif

// ---------------------------------------------------------------------------
// fp32 -> fp16 weight conversion
// ---------------------------------------------------------------------------
__global__ void k_f32_to_f16(const float* __restrict__ src, _Float16* __restrict__ dst, int n) {
    int i = blockIdx.x * blockDim.x + threadIdx.x;
    if (i < n) dst[i] = (_Float16)src[i];
}

#if HAVE_TDM
// ---------------------------------------------------------------------------
// Tensor Data Mover: async 2D tile load (fp32 elements) global -> LDS.
//   tile_dim0 = contiguous elements (pixels), tile_dim1 = lines (channels),
//   line stride = stride0 elements.  D# layout per CDNA5 ISA 8.3/8.4:
//   g0: [1:0]=count=1, [63:32]=lds_addr, [120:64]=global_addr, [127:126]=type 2
//   g1: [17:16]=data_size(2 -> 4B), [79:48]=tensor_dim0, [111:80]=tensor_dim1,
//       [127:112]=tile_dim0, [143:128]=tile_dim1, [207:160]=tensor_dim0_stride
// clang-23 toolchain: 6-arg builtin (g0, g1, g2, g3, g3ext, cpol)
// ---------------------------------------------------------------------------
__device__ __forceinline__ void tdm_load_2d_f32(unsigned lds_byte, const float* gptr,
                                                unsigned tile_d0, unsigned tile_d1,
                                                unsigned long long stride0)
{
    unsigned long long ga = (unsigned long long)(uintptr_t)gptr;
    u32x4 g0;
    g0[0] = 1u;                                              // count=1, user mode
    g0[1] = lds_byte;                                        // LDS byte address
    g0[2] = (unsigned)(ga & 0xffffffffull);
    g0[3] = (unsigned)((ga >> 32) & 0x1ffffffull) | (2u << 30); // type = 2 (image)

    unsigned td0 = tile_d0, td1 = tile_d1;                   // tensor dims == tile dims (in-bounds)
    i32x8 g1;
    g1[0] = (int)(2u << 16);                                 // data_size = 4 bytes
    g1[1] = (int)((td0 & 0xffffu) << 16);                    // tensor_dim0[15:0]
    g1[2] = (int)((td0 >> 16) | ((td1 & 0xffffu) << 16));    // tensor_dim0[31:16] | tensor_dim1[15:0]
    g1[3] = (int)((td1 >> 16) | (tile_d0 << 16));            // tensor_dim1[31:16] | tile_dim0
    g1[4] = (int)(tile_d1 & 0xffffu);                        // tile_dim1 (tile_dim2 = 0)
    g1[5] = (int)(stride0 & 0xffffffffull);                  // tensor_dim0_stride[31:0]
    g1[6] = (int)((stride0 >> 32) & 0xffffull);              // tensor_dim0_stride[47:32]
    g1[7] = 0;

    i32x4 z4 = {0, 0, 0, 0};
    i32x8 z8 = {0, 0, 0, 0, 0, 0, 0, 0};
    __builtin_amdgcn_tensor_load_to_lds(g0, g1, z4, z4, z8, 0);
}
__device__ __forceinline__ unsigned lds_byte_addr(const void* p) {
    // generic LDS address = {aperture, offset32}; low 32 bits are the LDS byte offset
    return (unsigned)((unsigned long long)(uintptr_t)p & 0xffffffffull);
}
#endif

// ---------------------------------------------------------------------------
// 1x1 stride-1 convolution == GEMM, TDM-staged, 4 WMMAs per K-chunk.
// One wave computes a 16(cout) x 64(pixel) tile.  Requires K%32==0, HW%64==0.
// B slab (32 channels x 64 pixels, fp32) is DMA'd into LDS by the Tensor Data
// Mover and double-buffered on TENSORcnt; A (weights, fp16) from global.
// ---------------------------------------------------------------------------
__global__ __launch_bounds__(32)
void k_conv1x1_wmma(const float* __restrict__ in, const _Float16* __restrict__ w16,
                    float* __restrict__ out, int K /*Cin*/, int Ctot, int c_off,
                    int HW, int Cout)
{
    const int lane = threadIdx.x;
    const int hi   = lane >> 4;
    const int lo   = lane & 15;

    const int slabsPerN = HW >> 6;                 // HW/64
    const int n   = blockIdx.x / slabsPerN;
    const int hw0 = (blockIdx.x % slabsPerN) << 6;

    const int co_base = blockIdx.y * 16;
    const int coA     = co_base + lo;
    const bool cvalid = (coA < Cout);
    const _Float16* wrow = w16 + (size_t)coA * K;

    const float* xbase = in + ((size_t)n * Ctot + c_off) * HW + hw0;
    const int nchunks = K >> 5;                    // K/32

    v8f acc0 = {}, acc1 = {}, acc2 = {}, acc3 = {};

#if HAVE_TDM
    __shared__ float lds[2][32 * 64];
    const unsigned loff0 = lds_byte_addr(&lds[0][0]);
    const unsigned loff1 = lds_byte_addr(&lds[1][0]);
    // prologue: DMA chunk 0 into buffer 0
    tdm_load_2d_f32(loff0, xbase, 64, 32, (unsigned long long)HW);
#endif

    for (int c = 0; c < nchunks; ++c) {
        const int kb = c << 5;
        if (kb + 32 < K) __builtin_prefetch(wrow + kb + 32, 0, 0);

        // ---- A: weights (fp16), contiguous k = 16*hi + i order, as B ----
        v16h a;
#pragma unroll
        for (int i = 0; i < 16; ++i) {
            int koff = i + 16 * hi;
            a[i] = cvalid ? wrow[kb + koff] : (_Float16)0.f;
        }

#if HAVE_TDM
        // pipeline: issue next DMA, then wait for the current buffer
        if (c + 1 < nchunks) {
            tdm_load_2d_f32((c & 1) ? loff0 : loff1,
                            xbase + (size_t)(c + 1) * 32 * HW, 64, 32,
                            (unsigned long long)HW);
            __builtin_amdgcn_s_wait_tensorcnt(1);
        } else {
            __builtin_amdgcn_s_wait_tensorcnt(0);
        }
        const float* L = lds[c & 1];
        const int krow = hi << 4;                  // this half-wave covers K rows 0..15 / 16..31
#pragma unroll
        for (int t = 0; t < 4; ++t) {
            v16h b;
            const int px = (t << 4) + lo;
#pragma unroll
            for (int i = 0; i < 16; ++i)
                b[i] = (_Float16)L[(krow + i) * 64 + px];
            if (t == 0) acc0 = __builtin_amdgcn_wmma_f32_16x16x32_f16(false, a, false, b, (short)0, acc0, false, false);
            if (t == 1) acc1 = __builtin_amdgcn_wmma_f32_16x16x32_f16(false, a, false, b, (short)0, acc1, false, false);
            if (t == 2) acc2 = __builtin_amdgcn_wmma_f32_16x16x32_f16(false, a, false, b, (short)0, acc2, false, false);
            if (t == 3) acc3 = __builtin_amdgcn_wmma_f32_16x16x32_f16(false, a, false, b, (short)0, acc3, false, false);
        }
#else
        const int krow = hi << 4;
#pragma unroll
        for (int t = 0; t < 4; ++t) {
            v16h b;
            const int px = (t << 4) + lo;
#pragma unroll
            for (int i = 0; i < 16; ++i)
                b[i] = (_Float16)xbase[(size_t)(kb + krow + i) * HW + px];
            if (t == 0) acc0 = __builtin_amdgcn_wmma_f32_16x16x32_f16(false, a, false, b, (short)0, acc0, false, false);
            if (t == 1) acc1 = __builtin_amdgcn_wmma_f32_16x16x32_f16(false, a, false, b, (short)0, acc1, false, false);
            if (t == 2) acc2 = __builtin_amdgcn_wmma_f32_16x16x32_f16(false, a, false, b, (short)0, acc2, false, false);
            if (t == 3) acc3 = __builtin_amdgcn_wmma_f32_16x16x32_f16(false, a, false, b, (short)0, acc3, false, false);
        }
#endif
    }

    float* obase = out + (size_t)n * Cout * HW + hw0;
#pragma unroll
    for (int r = 0; r < 8; ++r) {
        int co = co_base + r + 8 * hi;
        if (co < Cout) {
            float* orow = obase + (size_t)co * HW;
            orow[ 0 + lo] = acc0[r];
            orow[16 + lo] = acc1[r];
            orow[32 + lo] = acc2[r];
            orow[48 + lo] = acc3[r];
        }
    }
}

// ---------------------------------------------------------------------------
// Generic implicit-GEMM convolution via V_WMMA_F32_16X16X32_F16
// (7x7 / 3x3 / strided / small-K cases). One wave = 16(cout) x 16(pixel) tile.
// ---------------------------------------------------------------------------
__global__ __launch_bounds__(32)
void k_conv_wmma(const float* __restrict__ in, const _Float16* __restrict__ w16,
                 const float* __restrict__ bias, float* __restrict__ out,
                 int N, int Cin, int Ctot, int c_off, int H, int W,
                 int Cout, int KH, int KW, int stride, int pad, int Ho, int Wo)
{
    const int lane = threadIdx.x;
    const int hi   = lane >> 4;
    const int lo   = lane & 15;
    const int K    = Cin * KH * KW;
    const int P    = N * Ho * Wo;
    const int khkw = KH * KW;

    const int p  = blockIdx.x * 16 + lo;
    const bool pvalid = (p < P);
    int n = 0, oy = 0, ox = 0;
    if (pvalid) {
        n = p / (Ho * Wo);
        int rem = p - n * (Ho * Wo);
        oy = rem / Wo;
        ox = rem - oy * Wo;
    }

    const int co_base = blockIdx.y * 16;
    const int coA     = co_base + lo;
    const bool cvalid = (coA < Cout);
    const _Float16* wrow = w16 + (size_t)coA * K;

    v8f acc = {};

    for (int kb = 0; kb < K; kb += 32) {
        if (kb + 32 < K) __builtin_prefetch(wrow + kb + 32, 0, 0);

        v16h a;
#pragma unroll
        for (int i = 0; i < 16; ++i) {
            int koff = i + 16 * hi;
            int k = kb + koff;
            a[i] = (cvalid && k < K) ? wrow[k] : (_Float16)0.f;
        }

        v16h b;
        const int kstart = kb + (hi ? 16 : 0);
#pragma unroll
        for (int i = 0; i < 16; ++i) {
            int k = kstart + i;
            float v = 0.f;
            if (pvalid && k < K) {
                int ci = k / khkw;
                int t  = k - ci * khkw;
                int ky = t / KW;
                int kx = t - ky * KW;
                int iy = oy * stride - pad + ky;
                int ix = ox * stride - pad + kx;
                if (iy >= 0 && iy < H && ix >= 0 && ix < W)
                    v = in[(((size_t)n * Ctot + c_off + ci) * H + iy) * W + ix];
            }
            b[i] = (_Float16)v;
        }

        acc = __builtin_amdgcn_wmma_f32_16x16x32_f16(
                  false, a, false, b, (short)0, acc, false, false);
    }

    if (pvalid) {
#pragma unroll
        for (int r = 0; r < 8; ++r) {
            int co = co_base + r + 8 * hi;
            if (co < Cout) {
                float v = acc[r];
                if (bias) v += bias[co];
                out[(((size_t)n * Cout + co) * Ho + oy) * Wo + ox] = v;
            }
        }
    }
}

// ---------------------------------------------------------------------------
// Per-channel batch statistics: mean and 1/sqrt(var+eps) over (N,H,W)
// ---------------------------------------------------------------------------
__global__ void k_bn_stats(const float* __restrict__ x, float* __restrict__ mean,
                           float* __restrict__ istd, int C, int N, int HW, float eps)
{
    const int c = blockIdx.x;
    const int tid = threadIdx.x;
    const long total = (long)N * HW;
    float s = 0.f, s2 = 0.f;
    for (long i = tid; i < total; i += blockDim.x) {
        int nn = (int)(i / HW);
        int r  = (int)(i - (long)nn * HW);
        float v = x[((size_t)nn * C + c) * HW + r];
        s += v; s2 += v * v;
    }
    __shared__ float sh[256], sh2[256];
    sh[tid] = s; sh2[tid] = s2;
    __syncthreads();
    for (int o = 128; o > 0; o >>= 1) {
        if (tid < o) { sh[tid] += sh[tid + o]; sh2[tid] += sh2[tid + o]; }
        __syncthreads();
    }
    if (tid == 0) {
        float m = sh[0] / (float)total;
        float v = sh2[0] / (float)total - m * m;
        mean[c] = m;
        istd[c] = rsqrtf(v + eps);
    }
}

// mode: 0 = bn, 1 = bn+relu, 2 = bn + residual + relu, 3 = bn + tanh
__global__ void k_bn_apply(const float* __restrict__ x, const float* __restrict__ res,
                           float* __restrict__ y, const float* __restrict__ mean,
                           const float* __restrict__ istd, int C, int HW, long total, int mode)
{
    long i = (long)blockIdx.x * blockDim.x + threadIdx.x;
    if (i >= total) return;
    int c = (int)((i / HW) % C);
    float v = (x[i] - mean[c]) * istd[c];
    if (mode == 2) v += res[i];
    if (mode == 1 || mode == 2) v = fmaxf(v, 0.f);
    if (mode == 3) v = tanhf(v);
    y[i] = v;
}

__global__ void k_maxpool3s2(const float* __restrict__ x, float* __restrict__ y,
                             int NC, int H, int W, int Ho, int Wo)
{
    long i = (long)blockIdx.x * blockDim.x + threadIdx.x;
    long total = (long)NC * Ho * Wo;
    if (i >= total) return;
    int ox = (int)(i % Wo);
    int oy = (int)((i / Wo) % Ho);
    int nc = (int)(i / ((long)Ho * Wo));
    float m = -INFINITY;
    for (int dy = 0; dy < 3; ++dy) {
        int iy = oy * 2 - 1 + dy;
        if (iy < 0 || iy >= H) continue;
        for (int dx = 0; dx < 3; ++dx) {
            int ix = ox * 2 - 1 + dx;
            if (ix < 0 || ix >= W) continue;
            m = fmaxf(m, x[((size_t)nc * H + iy) * W + ix]);
        }
    }
    y[i] = m;
}

__global__ void k_avgpool2(const float* __restrict__ x, float* __restrict__ y,
                           int NC, int H, int W)
{
    int Ho = H / 2, Wo = W / 2;
    long i = (long)blockIdx.x * blockDim.x + threadIdx.x;
    long total = (long)NC * Ho * Wo;
    if (i >= total) return;
    int ox = (int)(i % Wo);
    int oy = (int)((i / Wo) % Ho);
    int nc = (int)(i / ((long)Ho * Wo));
    const float* p = x + ((size_t)nc * H + oy * 2) * W + ox * 2;
    y[i] = 0.25f * (p[0] + p[1] + p[W] + p[W + 1]);
}

__global__ void k_chan_copy(const float* __restrict__ src, float* __restrict__ dst,
                            int N, int Cs, int Cd, int coff, int HW)
{
    long i = (long)blockIdx.x * blockDim.x + threadIdx.x;
    long total = (long)N * Cs * HW;
    if (i >= total) return;
    int q = (int)(i % HW);
    int c = (int)((i / HW) % Cs);
    int n = (int)(i / ((long)Cs * HW));
    dst[((size_t)n * Cd + coff + c) * HW + q] = src[i];
}

__global__ void k_grid_sample_acc(const float* __restrict__ img, const float* __restrict__ grid,
                                  float* __restrict__ outv, float* __restrict__ sum,
                                  float sign, float gscale, int N, int C, int H, int W)
{
    long i = (long)blockIdx.x * blockDim.x + threadIdx.x;
    long total = (long)N * C * H * W;
    if (i >= total) return;
    int HW = H * W;
    int w = (int)(i % W);
    int h = (int)((i / W) % H);
    int c = (int)((i / HW) % C);
    int n = (int)(i / ((long)C * HW));

    float gx = sign * gscale * grid[(((size_t)n * 2 + 0) * H + h) * W + w];
    float gy = sign * gscale * grid[(((size_t)n * 2 + 1) * H + h) * W + w];
    float fx = (gx + 1.f) * (W * 0.5f) - 0.5f;
    float fy = (gy + 1.f) * (H * 0.5f) - 0.5f;
    float x0f = floorf(fx), y0f = floorf(fy);
    float wx = fx - x0f, wy = fy - y0f;
    int x0 = (int)x0f, y0 = (int)y0f;

    const float* imnc = img + ((size_t)n * C + c) * HW;
    float acc = 0.f;
#pragma unroll
    for (int dy = 0; dy < 2; ++dy) {
#pragma unroll
        for (int dx = 0; dx < 2; ++dx) {
            int xi = x0 + dx, yi = y0 + dy;
            float wgt = (dx ? wx : 1.f - wx) * (dy ? wy : 1.f - wy);
            if (xi >= 0 && xi < W && yi >= 0 && yi < H)
                acc += wgt * imnc[yi * W + xi];
        }
    }
    outv[i] = acc;
    sum[i] += acc;
}

// ---------------------------------------------------------------------------
// Host orchestration
// ---------------------------------------------------------------------------
extern "C" void kernel_launch(void* const* d_in, const int* in_sizes, int n_in,
                              void* d_out, int out_size, void* d_ws, size_t ws_size,
                              hipStream_t stream)
{
    const int N = 2;
    const int HW64 = 64 * 64;

    char* base = (char*)d_ws;
    size_t off = 0;
    auto alloc = [&](size_t bytes) -> char* {
        off = (off + 255) & ~(size_t)255;
        char* p = base + off;
        off += bytes;
        return p;
    };

    _Float16* wpool = (_Float16*)alloc(1310720);
    float* mean = (float*)alloc(768 * sizeof(float));
    float* istd = (float*)alloc(768 * sizeof(float));

    float* buf256A = (float*)alloc((size_t)2 * 64 * 256 * 256 * 4);
    float* buf256B = (float*)alloc((size_t)2 * 64 * 256 * 256 * 4);
    float* buf128A = (float*)alloc((size_t)2 * 64 * 128 * 128 * 4);
    float* buf128B = (float*)alloc((size_t)2 * 64 * 128 * 128 * 4);
    float* b64a    = (float*)alloc((size_t)2 * 64 * HW64 * 4);
    float* b64b    = (float*)alloc((size_t)2 * 64 * HW64 * 4);
    float* C0      = (float*)alloc((size_t)2 * 256 * HW64 * 4);
    float* C1      = (float*)alloc((size_t)2 * 256 * HW64 * 4);
    float* C2      = (float*)alloc((size_t)2 * 256 * HW64 * 4);
    float* dg128   = (float*)alloc((size_t)2 * 2 * 128 * 128 * 4);
    float* dg64    = (float*)alloc((size_t)2 * 2 * HW64 * 4);
    float* feat2   = (float*)alloc((size_t)2 * 2 * HW64 * 4);
    float* g4      = (float*)alloc((size_t)2 * 4 * HW64 * 4);
    float* g8a     = (float*)alloc((size_t)2 * 8 * HW64 * 4);
    float* g8b     = (float*)alloc((size_t)2 * 8 * HW64 * 4);
    float* g32a    = (float*)alloc((size_t)2 * 32 * HW64 * 4);
    float* g32b    = (float*)alloc((size_t)2 * 32 * HW64 * 4);
    float* g32c    = (float*)alloc((size_t)2 * 32 * HW64 * 4);
    float* gridb   = (float*)alloc((size_t)2 * 2 * HW64 * 4);

    const size_t ELT256 = (size_t)2 * 256 * HW64;
    float* o1   = buf256A;
    float* o2   = buf256A + ELT256;
    float* s1   = buf256A + 2 * ELT256;
    float* s2   = buf256A + 3 * ELT256;
    float* spare = buf256B;
    float* cat768 = buf256B + ELT256;

    size_t wOff = 0;
    auto cvt = [&](int idx, int n) -> const _Float16* {
        _Float16* dst = wpool + wOff;
        wOff += (size_t)((n + 7) & ~7);
        k_f32_to_f16<<<DIVUP(n, 256), 256, 0, stream>>>((const float*)d_in[idx], dst, n);
        return dst;
    };
    const _Float16* w_conv1 = cvt(1, 6272);
    const _Float16* w_conv2 = cvt(4, 36864);
    const _Float16* l2b0_c1 = cvt(7, 4096);
    const _Float16* l2b0_c2 = cvt(10, 36864);
    const _Float16* l2b0_c3 = cvt(13, 16384);
    const _Float16* l2b0_dn = cvt(16, 16384);
    const _Float16* l2b1_c1 = cvt(19, 16384);
    const _Float16* l2b1_c2 = cvt(22, 36864);
    const _Float16* l2b1_c3 = cvt(25, 16384);
    const _Float16* l2b2_c1 = cvt(28, 16384);
    const _Float16* l2b2_c2 = cvt(31, 36864);
    const _Float16* l2b2_c3 = cvt(34, 16384);
    const _Float16* w_dem   = cvt(37, 18);
    const float*    b_dem   = (const float*)d_in[38];
    const _Float16* w_fs    = cvt(39, 512);
    const _Float16* g0_c1   = cvt(40, 32);
    const _Float16* g0_c2   = cvt(43, 576);
    const _Float16* g0_c3   = cvt(46, 256);
    const _Float16* g0_dn   = cvt(49, 128);
    const _Float16* g1_c1   = cvt(52, 256);
    const _Float16* g1_c2   = cvt(55, 576);
    const _Float16* g1_c3   = cvt(58, 256);
    const _Float16* g2_c1   = cvt(61, 256);
    const _Float16* g2_c2   = cvt(64, 576);
    const _Float16* g2_c3   = cvt(67, 256);
    const _Float16* w_gsq   = cvt(70, 64);
    const _Float16* w_merge = cvt(73, 196608);
    const _Float16* pp0_c1  = cvt(74, 16384);
    const _Float16* pp0_c2  = cvt(77, 36864);
    const _Float16* pp0_c3  = cvt(80, 16384);
    const _Float16* pp1_c1  = cvt(83, 16384);
    const _Float16* pp1_c2  = cvt(86, 36864);
    const _Float16* pp1_c3  = cvt(89, 16384);

    auto conv = [&](const float* in, int c_off, int Ctot, const _Float16* w,
                    const float* bias, float* out, int Cin, int H, int W,
                    int Cout, int KH, int KW, int stride, int pad, int Ho, int Wo) {
        int HW = H * W;
        if (KH == 1 && KW == 1 && stride == 1 && pad == 0 && bias == nullptr &&
            (Cin % 32) == 0 && (HW % 64) == 0) {
            dim3 grid(N * (HW / 64), DIVUP(Cout, 16));
            k_conv1x1_wmma<<<grid, 32, 0, stream>>>(in, w, out, Cin, Ctot, c_off, HW, Cout);
        } else {
            dim3 grid(DIVUP(N * Ho * Wo, 16), DIVUP(Cout, 16));
            k_conv_wmma<<<grid, 32, 0, stream>>>(in, w, bias, out, N, Cin, Ctot, c_off,
                                                 H, W, Cout, KH, KW, stride, pad, Ho, Wo);
        }
    };
    auto bnstats = [&](const float* x, int C, int HW) {
        k_bn_stats<<<C, 256, 0, stream>>>(x, mean, istd, C, N, HW, 1e-5f);
    };
    auto bnapply = [&](const float* x, const float* res, float* y, int C, int HW, int mode) {
        long total = (long)N * C * HW;
        k_bn_apply<<<DIVUP(total, 256), 256, 0, stream>>>(x, res, y, mean, istd, C, HW, total, mode);
    };
    auto bnrelu = [&](float* x, int C, int HW) { bnstats(x, C, HW); bnapply(x, nullptr, x, C, HW, 1); };

    const float* xin = (const float*)d_in[0];

    // ---- stem ----
    conv(xin, 0, 3, w_conv1, nullptr, buf256A, 2, 256, 256, 64, 7, 7, 1, 3, 256, 256);
    bnrelu(buf256A, 64, 65536);
    conv(buf256A, 0, 64, w_conv2, nullptr, buf256B, 64, 256, 256, 64, 3, 3, 1, 1, 256, 256);
    bnrelu(buf256B, 64, 65536);
    {
        long total = (long)N * 64 * 128 * 128;
        k_maxpool3s2<<<DIVUP(total, 256), 256, 0, stream>>>(buf256B, buf128A, N * 64, 256, 256, 128, 128);
    }

    // ---- layer2 block0 (stride 2, down) ----
    conv(buf128A, 0, 64, l2b0_c1, nullptr, buf128B, 64, 128, 128, 64, 1, 1, 1, 0, 128, 128);
    bnrelu(buf128B, 64, 128 * 128);
    conv(buf128B, 0, 64, l2b0_c2, nullptr, b64a, 64, 128, 128, 64, 3, 3, 2, 1, 64, 64);
    bnrelu(b64a, 64, HW64);
    conv(b64a, 0, 64, l2b0_c3, nullptr, C0, 64, 64, 64, 256, 1, 1, 1, 0, 64, 64);
    conv(buf128A, 0, 64, l2b0_dn, nullptr, C1, 64, 128, 128, 256, 1, 1, 2, 0, 64, 64);
    bnstats(C1, 256, HW64); bnapply(C1, nullptr, C1, 256, HW64, 0);
    bnstats(C0, 256, HW64); bnapply(C0, C1, C2, 256, HW64, 2);       // x = C2

    // ---- layer2 block1 ----
    conv(C2, 0, 256, l2b1_c1, nullptr, b64a, 256, 64, 64, 64, 1, 1, 1, 0, 64, 64);
    bnrelu(b64a, 64, HW64);
    conv(b64a, 0, 64, l2b1_c2, nullptr, b64b, 64, 64, 64, 64, 3, 3, 1, 1, 64, 64);
    bnrelu(b64b, 64, HW64);
    conv(b64b, 0, 64, l2b1_c3, nullptr, C0, 64, 64, 64, 256, 1, 1, 1, 0, 64, 64);
    bnstats(C0, 256, HW64); bnapply(C0, C2, C1, 256, HW64, 2);       // x = C1

    // ---- layer2 block2 ----
    conv(C1, 0, 256, l2b2_c1, nullptr, b64a, 256, 64, 64, 64, 1, 1, 1, 0, 64, 64);
    bnrelu(b64a, 64, HW64);
    conv(b64a, 0, 64, l2b2_c2, nullptr, b64b, 64, 64, 64, 64, 3, 3, 1, 1, 64, 64);
    bnrelu(b64b, 64, HW64);
    conv(b64b, 0, 64, l2b2_c3, nullptr, C0, 64, 64, 64, 256, 1, 1, 1, 0, 64, 64);
    bnstats(C0, 256, HW64); bnapply(C0, C1, C2, 256, HW64, 2);       // x = C2
    float* xfeat = C2;

    // ---- DEM gradients path ----
    conv(xin, 2, 3, w_dem, b_dem, dg128, 1, 256, 256, 2, 3, 3, 2, 1, 128, 128);
    {
        long total = (long)N * 2 * HW64;
        k_avgpool2<<<DIVUP(total, 256), 256, 0, stream>>>(dg128, dg64, N * 2, 128, 128);
    }
    conv(xfeat, 0, 256, w_fs, nullptr, feat2, 256, 64, 64, 2, 1, 1, 1, 0, 64, 64);
    {
        long t2 = (long)N * 2 * HW64;
        k_chan_copy<<<DIVUP(t2, 256), 256, 0, stream>>>(dg64, g4, N, 2, 4, 0, HW64);
        k_chan_copy<<<DIVUP(t2, 256), 256, 0, stream>>>(feat2, g4, N, 2, 4, 2, HW64);
    }

    // ---- grid layer block0 (down, stride 1) ----
    conv(g4, 0, 4, g0_c1, nullptr, g8a, 4, 64, 64, 8, 1, 1, 1, 0, 64, 64);
    bnrelu(g8a, 8, HW64);
    conv(g8a, 0, 8, g0_c2, nullptr, g8b, 8, 64, 64, 8, 3, 3, 1, 1, 64, 64);
    bnrelu(g8b, 8, HW64);
    conv(g8b, 0, 8, g0_c3, nullptr, g32a, 8, 64, 64, 32, 1, 1, 1, 0, 64, 64);
    conv(g4, 0, 4, g0_dn, nullptr, g32b, 4, 64, 64, 32, 1, 1, 1, 0, 64, 64);
    bnstats(g32b, 32, HW64); bnapply(g32b, nullptr, g32b, 32, HW64, 0);
    bnstats(g32a, 32, HW64); bnapply(g32a, g32b, g32c, 32, HW64, 2); // g = g32c

    // ---- grid layer block1 ----
    conv(g32c, 0, 32, g1_c1, nullptr, g8a, 32, 64, 64, 8, 1, 1, 1, 0, 64, 64);
    bnrelu(g8a, 8, HW64);
    conv(g8a, 0, 8, g1_c2, nullptr, g8b, 8, 64, 64, 8, 3, 3, 1, 1, 64, 64);
    bnrelu(g8b, 8, HW64);
    conv(g8b, 0, 8, g1_c3, nullptr, g32a, 8, 64, 64, 32, 1, 1, 1, 0, 64, 64);
    bnstats(g32a, 32, HW64); bnapply(g32a, g32c, g32b, 32, HW64, 2); // g = g32b

    // ---- grid layer block2 ----
    conv(g32b, 0, 32, g2_c1, nullptr, g8a, 32, 64, 64, 8, 1, 1, 1, 0, 64, 64);
    bnrelu(g8a, 8, HW64);
    conv(g8a, 0, 8, g2_c2, nullptr, g8b, 8, 64, 64, 8, 3, 3, 1, 1, 64, 64);
    bnrelu(g8b, 8, HW64);
    conv(g8b, 0, 8, g2_c3, nullptr, g32a, 8, 64, 64, 32, 1, 1, 1, 0, 64, 64);
    bnstats(g32a, 32, HW64); bnapply(g32a, g32b, g32c, 32, HW64, 2); // g = g32c

    // ---- grid squeeze (scale H/64 == 1.0, folded into sampler) ----
    conv(g32c, 0, 32, w_gsq, nullptr, gridb, 32, 64, 64, 2, 1, 1, 1, 0, 64, 64);
    const float gscale = 64.f / 64.f;

    // ---- 50-iteration grid-sample scan ----
    const size_t xbytes = ELT256 * sizeof(float);
    (void)hipMemcpyAsync(o1, xfeat, xbytes, hipMemcpyDeviceToDevice, stream);
    (void)hipMemcpyAsync(o2, xfeat, xbytes, hipMemcpyDeviceToDevice, stream);
    (void)hipMemsetAsync(s1, 0, xbytes, stream);
    (void)hipMemsetAsync(s2, 0, xbytes, stream);
    {
        long total = (long)ELT256;
        int blocks = (int)DIVUP(total, 256);
        for (int it = 0; it < 50; ++it) {
            k_grid_sample_acc<<<blocks, 256, 0, stream>>>(o1, gridb, spare, s1, +1.f, gscale, N, 256, 64, 64);
            std::swap(o1, spare);
            k_grid_sample_acc<<<blocks, 256, 0, stream>>>(o2, gridb, spare, s2, -1.f, gscale, N, 256, 64, 64);
            std::swap(o2, spare);
        }
    }

    // ---- concat [x, sum1, sum2] -> (2,768,64,64), BN + tanh ----
    {
        long t = (long)N * 256 * HW64;
        int blocks = (int)DIVUP(t, 256);
        k_chan_copy<<<blocks, 256, 0, stream>>>(xfeat, cat768, N, 256, 768, 0, HW64);
        k_chan_copy<<<blocks, 256, 0, stream>>>(s1,    cat768, N, 256, 768, 256, HW64);
        k_chan_copy<<<blocks, 256, 0, stream>>>(s2,    cat768, N, 256, 768, 512, HW64);
    }
    bnstats(cat768, 768, HW64);
    bnapply(cat768, nullptr, cat768, 768, HW64, 3);

    // ---- merge 768 -> 256 ----
    conv(cat768, 0, 768, w_merge, nullptr, C0, 768, 64, 64, 256, 1, 1, 1, 0, 64, 64);

    // ---- postprocess block0 ----
    conv(C0, 0, 256, pp0_c1, nullptr, b64a, 256, 64, 64, 64, 1, 1, 1, 0, 64, 64);
    bnrelu(b64a, 64, HW64);
    conv(b64a, 0, 64, pp0_c2, nullptr, b64b, 64, 64, 64, 64, 3, 3, 1, 1, 64, 64);
    bnrelu(b64b, 64, HW64);
    conv(b64b, 0, 64, pp0_c3, nullptr, C2, 64, 64, 64, 256, 1, 1, 1, 0, 64, 64);
    bnstats(C2, 256, HW64); bnapply(C2, C0, C1, 256, HW64, 2);       // x = C1

    // ---- postprocess block1 -> d_out ----
    conv(C1, 0, 256, pp1_c1, nullptr, b64a, 256, 64, 64, 64, 1, 1, 1, 0, 64, 64);
    bnrelu(b64a, 64, HW64);
    conv(b64a, 0, 64, pp1_c2, nullptr, b64b, 64, 64, 64, 64, 3, 3, 1, 1, 64, 64);
    bnrelu(b64b, 64, HW64);
    conv(b64b, 0, 64, pp1_c3, nullptr, C2, 64, 64, 64, 256, 1, 1, 1, 0, 64, 64);
    bnstats(C2, 256, HW64);
    bnapply(C2, C1, (float*)d_out, 256, HW64, 2);

    (void)in_sizes; (void)n_in; (void)out_size; (void)ws_size;
}
